// PiecewiseRationalQuadraticCouplingTransform_58832462021178
// MI455X (gfx1250) — matrix operand order repaired and checked by reading, not compiled, not run on hardware
//
#include <hip/hip_runtime.h>
#include <hip/hip_bf16.h>

// ---------------- types ----------------
typedef __attribute__((ext_vector_type(16))) _Float16 v16h;
typedef __attribute__((ext_vector_type(8)))  _Float16 v8h;
typedef __attribute__((ext_vector_type(8)))  float    v8f;

#define BATCH    8192
#define FEAT     512
#define NTR      256     // transformed features
#define HIDDEN   1024
#define OUTDIM   12032   // 256 * 47
#define NPAD     12064   // OUTDIM padded to 32 for transposed W2
#define KBINS    16
#define TBOUND   3.0f
#define MBW      0.001f
#define MBH      0.001f
#define MDERIV   0.001f
#define DCONST   0.53974175f   // log(exp(0.999)-1)

// ---- A fragment: 16x32 f16, per ISA table (khalf interleave) ----
__device__ __forceinline__ v16h load_fragA(const _Float16* rowbase, int khalf) {
    v8h lo = *(const v8h*)(rowbase + khalf);        // K = khalf + 0..7
    v8h hi = *(const v8h*)(rowbase + 16 + khalf);   // K = 16 + khalf + 0..7
    v16h r;
#pragma unroll
    for (int i = 0; i < 8; ++i) { r[i] = lo[i]; r[8 + i] = hi[i]; }
    return r;
}
// ---- B fragment: 32x16 f16, per ISA table (koff = khalf interleave) ----
__device__ __forceinline__ v16h load_fragB(const _Float16* colbase, int koff) {
    v8h lo = *(const v8h*)(colbase + koff);
    v8h hi = *(const v8h*)(colbase + koff + 16);
    v16h r;
#pragma unroll
    for (int i = 0; i < 8; ++i) { r[i] = lo[i]; r[8 + i] = hi[i]; }
    return r;
}

// ---- gfx1250 async global->LDS copy (ASYNCcnt-tracked, bypasses VGPRs) ----
__device__ __forceinline__ unsigned lds_off(const void* p) {
    return (unsigned)(unsigned long long)p;   // low 32 bits = LDS byte address
}
__device__ __forceinline__ void async_copy_b128(unsigned lds_addr, const void* gaddr) {
    asm volatile("global_load_async_to_lds_b128 %0, %1, off"
                 :: "v"(lds_addr), "v"(gaddr) : "memory");
}
__device__ __forceinline__ void wait_async0() {
    asm volatile("s_wait_asynccnt 0x0" ::: "memory");
}

// ---------------- prep kernels (one-off, bandwidth-trivial) ----------------
// identity columns -> packed f16 matrix; also identity passthrough to output
__global__ void make_identity_f16(const float* __restrict__ in,
                                  _Float16* __restrict__ I16h,
                                  float* __restrict__ out) {
    int i = blockIdx.x * 256 + threadIdx.x;     // 8192*256
    if (i < BATCH * NTR) {
        int b = i >> 8, f = i & 255;
        float v = in[b * FEAT + 2 * f + 1];     // identity_idx = odd columns
        I16h[b * NTR + f] = (_Float16)v;
        out[b * FEAT + 2 * f + 1] = v;
    }
}

// LDS-tiled transpose + f32->f16: dst[n*K + k] = src[k*N + n]; pad rows zeroed
__global__ __launch_bounds__(256) void transpose_f32_to_f16(
    const float* __restrict__ src, _Float16* __restrict__ dst,
    int K, int N, int Npad) {
    __shared__ float tile[32][33];
    const int n0 = blockIdx.x * 32, k0 = blockIdx.y * 32;
    const int tx = threadIdx.x & 31, ty = threadIdx.x >> 5;   // 32 x 8
#pragma unroll
    for (int r = 0; r < 32; r += 8) {
        int k = k0 + ty + r, n = n0 + tx;
        tile[ty + r][tx] = (k < K && n < N) ? src[k * N + n] : 0.f;
    }
    __syncthreads();
#pragma unroll
    for (int r = 0; r < 32; r += 8) {
        int n = n0 + ty + r, k = k0 + tx;
        if (n < Npad) dst[n * K + k] = (_Float16)tile[tx][ty + r];
    }
}

// ------------- GEMM1: H = relu(I16h @ W1 + b1)  (async + WMMA) -------------
__global__ __launch_bounds__(256) void gemm1_relu(
    const _Float16* __restrict__ I16h,  // [8192 x 256] f16
    const _Float16* __restrict__ W1t,   // [1024 x 256] f16 (N-major)
    const float*    __restrict__ b1,    // [1024]
    _Float16*       __restrict__ H)     // [8192 x 1024]
{
    __shared__ __align__(16) _Float16 As[2][128 * 32];
    __shared__ __align__(16) _Float16 Bs[2][64 * 32];   // Bs[n][k]
    const int row0 = blockIdx.x * 128;
    const int col0 = blockIdx.y * 64;
    const int tid  = threadIdx.x;
    const int wave = tid >> 5;
    const int lane = tid & 31;
    const int lrow = lane & 15;
    const int khalf = (lane >> 4) * 8;
    const int koffB = (lane >> 4) * 8;

    // A: 512 chunks of 8 halfs, 2 per thread
    const int am0 = tid >> 2,          ak0 = (tid & 3) * 8;
    const int am1 = (tid + 256) >> 2,  ak1 = (tid & 3) * 8;
    const _Float16* gA0 = I16h + (row0 + am0) * NTR + ak0;
    const _Float16* gA1 = I16h + (row0 + am1) * NTR + ak1;
    const unsigned lA0[2] = { lds_off(&As[0][am0*32+ak0]), lds_off(&As[1][am0*32+ak0]) };
    const unsigned lA1[2] = { lds_off(&As[0][am1*32+ak1]), lds_off(&As[1][am1*32+ak1]) };
    // B: 256 chunks, 1 per thread
    const int bn = tid >> 2, bkk = (tid & 3) * 8;
    const _Float16* gB = W1t + (col0 + bn) * NTR + bkk;
    const unsigned lB[2] = { lds_off(&Bs[0][bn*32+bkk]), lds_off(&Bs[1][bn*32+bkk]) };

    v8f acc[4] = {};
    async_copy_b128(lA0[0], gA0);
    async_copy_b128(lA1[0], gA1);
    async_copy_b128(lB[0],  gB);

#pragma unroll 2
    for (int s = 0; s < 8; ++s) {
        const int cur = s & 1;
        wait_async0();
        __syncthreads();
        if (s + 1 < 8) {
            const int k0 = (s + 1) * 32;
            async_copy_b128(cur ? lA0[0] : lA0[1], gA0 + k0);
            async_copy_b128(cur ? lA1[0] : lA1[1], gA1 + k0);
            async_copy_b128(cur ? lB[0]  : lB[1],  gB  + k0);
        }
        v16h a = load_fragA(&As[cur][(wave * 16 + lrow) * 32], khalf);
#pragma unroll
        for (int j = 0; j < 4; ++j) {
            v16h b = load_fragB(&Bs[cur][(j * 16 + lrow) * 32], koffB);
            acc[j] = __builtin_amdgcn_wmma_f32_16x16x32_f16(
                false, a, false, b, (short)0, acc[j], false, false);
        }
    }
    // epilogue: bias + relu + f16 store
    const int rbase = row0 + wave * 16 + ((lane >> 4) ? 8 : 0);
#pragma unroll
    for (int j = 0; j < 4; ++j) {
        int col = col0 + j * 16 + lrow;
        float bias = b1[col];
#pragma unroll
        for (int r = 0; r < 8; ++r) {
            float v = acc[j][r] + bias;
            H[(rbase + r) * HIDDEN + col] = (_Float16)(v > 0.f ? v : 0.f);
        }
    }
}

// ------ GEMM2 fused with RQ spline: block = 128 rows x 2 features (N=96) ----
#define PSTR 97
__global__ __launch_bounds__(256) void gemm2_spline(
    const _Float16* __restrict__ H,     // [8192 x 1024] f16
    const _Float16* __restrict__ W2t,   // [12064 x 1024] f16 (N-major, pad zeroed)
    const float*    __restrict__ b2,    // [12032]
    const float*    __restrict__ inp,   // [8192 x 512]
    float*          __restrict__ out,   // [8192 x 512]
    float*          __restrict__ ladbuf)// [256 x 8192] feature-major partials
{
    __shared__ __align__(16) char smem[128 * PSTR * 4];   // 49664 B
    _Float16* AsB0 = (_Float16*)(smem);            // 2 x 8192 B
    _Float16* AsB1 = (_Float16*)(smem + 8192);
    _Float16* BsB0 = (_Float16*)(smem + 16384);    // 2 x 6144 B (Bs[n][k])
    _Float16* BsB1 = (_Float16*)(smem + 16384 + 6144);
    float*    P    = (float*)smem;                 // [128][97] after mainloop

    const int row0 = blockIdx.x * 128;
    const int t0   = blockIdx.y * 2;        // two features per block
    const int n0   = t0 * 47;               // 94 real param columns + 2 pad
    const int tid  = threadIdx.x;
    const int wave = tid >> 5;
    const int lane = tid & 31;
    const int lrow = lane & 15;
    const int khalf = (lane >> 4) * 8;
    const int koffB = (lane >> 4) * 8;

    // A: 512 chunks of 8 halfs, 2 per thread (async)
    const int am0 = tid >> 2,         ak0 = (tid & 3) * 8;
    const int am1 = (tid + 256) >> 2, ak1 = (tid & 3) * 8;
    const _Float16* gA0 = H + (row0 + am0) * HIDDEN + ak0;
    const _Float16* gA1 = H + (row0 + am1) * HIDDEN + ak1;
    const unsigned lA0[2] = { lds_off(AsB0 + am0*32 + ak0), lds_off(AsB1 + am0*32 + ak0) };
    const unsigned lA1[2] = { lds_off(AsB0 + am1*32 + ak1), lds_off(AsB1 + am1*32 + ak1) };
    // B: 96x32 halfs = 384 chunks; 1 per thread + 1 extra for tid<128 (async)
    const int bn0 = tid >> 2,         bk0 = (tid & 3) * 8;
    const int bn1 = (tid + 256) >> 2, bk1 = (tid & 3) * 8;
    const _Float16* gB0 = W2t + (n0 + bn0) * HIDDEN + bk0;
    const _Float16* gB1 = W2t + (n0 + bn1) * HIDDEN + bk1;
    const unsigned lB0[2] = { lds_off(BsB0 + bn0*32 + bk0), lds_off(BsB1 + bn0*32 + bk0) };
    const unsigned lB1[2] = { lds_off(BsB0 + bn1*32 + bk1), lds_off(BsB1 + bn1*32 + bk1) };
    const bool hasB1 = tid < 128;

    v8f acc[6] = {};

    // prologue: stage K-step 0 into buffer 0
    async_copy_b128(lA0[0], gA0);
    async_copy_b128(lA1[0], gA1);
    async_copy_b128(lB0[0], gB0);
    if (hasB1) async_copy_b128(lB1[0], gB1);

#pragma unroll 2
    for (int s = 0; s < 32; ++s) {
        const int cur = s & 1;
        wait_async0();
        __syncthreads();
        if (s + 1 < 32) {                       // stage next step into other buffer
            const int k0 = (s + 1) * 32;
            async_copy_b128(cur ? lA0[0] : lA0[1], gA0 + k0);
            async_copy_b128(cur ? lA1[0] : lA1[1], gA1 + k0);
            async_copy_b128(cur ? lB0[0] : lB0[1], gB0 + k0);
            if (hasB1) async_copy_b128(cur ? lB1[0] : lB1[1], gB1 + k0);
        }
        const _Float16* Ac = cur ? AsB1 : AsB0;
        const _Float16* Bc = cur ? BsB1 : BsB0;
        v16h a = load_fragA(Ac + (wave * 16 + lrow) * 32, khalf);
#pragma unroll
        for (int j = 0; j < 6; ++j) {
            v16h b = load_fragB(Bc + (j * 16 + lrow) * 32, koffB);
            acc[j] = __builtin_amdgcn_wmma_f32_16x16x32_f16(
                false, a, false, b, (short)0, acc[j], false, false);
        }
    }
    __syncthreads();                            // P aliases staging buffers

    // ---- dump C (+bias) to LDS params buffer ----
    const int rb = wave * 16 + ((lane >> 4) ? 8 : 0);
#pragma unroll
    for (int j = 0; j < 6; ++j) {
        int col = j * 16 + lrow;                // 0..95
        float bias = (col < 94) ? b2[n0 + col] : 0.f;
#pragma unroll
        for (int r = 0; r < 8; ++r)
            P[(rb + r) * PSTR + col] = acc[j][r] + bias;
    }
    __syncthreads();

    // -------- spline: 256 threads = 128 rows x 2 features, params in LDS ----
    {
        const int f = tid >> 7;
        const int r = tid & 127;
        const int t = t0 + f;
        const float* p = &P[r * PSTR + f * 47]; // [0,16)=uw [16,32)=uh [32,47)=ud
        const int grow = row0 + r;
        const float x = inp[grow * FEAT + 2 * t];   // transform_idx = even cols
        const bool inside = (x >= -TBOUND) && (x <= TBOUND);
        float xc = fminf(fmaxf(x, -TBOUND), TBOUND);

        float mw = -1e30f, mh = -1e30f;
#pragma unroll
        for (int i = 0; i < KBINS; ++i) {
            mw = fmaxf(mw, p[i]);
            mh = fmaxf(mh, p[16 + i]);
        }
        float sw = 0.f, sh = 0.f;
#pragma unroll
        for (int i = 0; i < KBINS; ++i) {
            sw += __expf(p[i] - mw);
            sh += __expf(p[16 + i] - mh);
        }
        const float rw = 1.f / sw, rh = 1.f / sh;
        const float scw = 1.f - MBW * KBINS, sch = 1.f - MBH * KBINS;

        float cw = -TBOUND, ch = -TBOUND, accw = 0.f, acch = 0.f;
        float in_cw = -TBOUND, in_w = 1.f, in_ch = -TBOUND, in_h = 1.f;
        float ud0 = DCONST, ud1 = DCONST;
#pragma unroll
        for (int i = 0; i < KBINS; ++i) {
            float wi = MBW + scw * (__expf(p[i] - mw) * rw);
            float hi = MBH + sch * (__expf(p[16 + i] - mh) * rh);
            accw += wi; acch += hi;
            float cwn = (i == KBINS - 1) ? TBOUND : (2.f * TBOUND * accw - TBOUND);
            float chn = (i == KBINS - 1) ? TBOUND : (2.f * TBOUND * acch - TBOUND);
            if (xc >= cw) {
                in_cw = cw; in_w = cwn - cw;
                in_ch = ch; in_h = chn - ch;
                ud0 = (i == 0) ? DCONST : p[31 + i];            // ud_full[i]
                ud1 = (i == KBINS - 1) ? DCONST : p[32 + i];    // ud_full[i+1]
            }
            cw = cwn; ch = chn;
        }
        float d0 = MDERIV + ((ud0 > 20.f) ? ud0 : log1pf(__expf(ud0)));
        float d1 = MDERIV + ((ud1 > 20.f) ? ud1 : log1pf(__expf(ud1)));
        float delta = in_h / in_w;
        float theta = (xc - in_cw) / in_w;
        float t1m = theta * (1.f - theta);
        float denom = delta + (d0 + d1 - 2.f * delta) * t1m;
        float numer = in_h * (delta * theta * theta + d0 * t1m);
        float y = in_ch + numer / denom;
        float omt = 1.f - theta;
        float dnum = delta * delta * (d1 * theta * theta + 2.f * delta * t1m + d0 * omt * omt);
        float ladv = __logf(dnum) - 2.f * __logf(denom);
        if (!inside) { y = x; ladv = 0.f; }
        out[grow * FEAT + 2 * t] = y;
        ladbuf[t * BATCH + grow] = ladv;        // deterministic (no atomics)
    }
}

// ---------------- deterministic logabsdet reduction ----------------
__global__ void reduce_lad(const float* __restrict__ lb, float* __restrict__ lad) {
    int r = blockIdx.x * 256 + threadIdx.x;     // 8192 rows
    if (r < BATCH) {
        float s = 0.f;
#pragma unroll 4
        for (int t = 0; t < NTR; ++t) s += lb[t * BATCH + r];  // coalesced
        lad[r] = s;
    }
}

// ---------------- launcher ----------------
extern "C" void kernel_launch(void* const* d_in, const int* in_sizes, int n_in,
                              void* d_out, int out_size, void* d_ws, size_t ws_size,
                              hipStream_t stream) {
    const float* inp = (const float*)d_in[0];   // [8192,512]
    const float* W1  = (const float*)d_in[1];   // [256,1024]
    const float* b1  = (const float*)d_in[2];   // [1024]
    const float* W2  = (const float*)d_in[3];   // [1024,12032]
    const float* b2  = (const float*)d_in[4];   // [12032]

    char* ws = (char*)d_ws;
    _Float16* I16h  = (_Float16*)(ws);                      // 4 MB   [8192 x 256]
    _Float16* W1t   = (_Float16*)(ws + 4194304);            // 512 KB [1024 x 256]
    _Float16* W2t   = (_Float16*)(ws + 4718592);            // 23.6MB [12064 x 1024]
    _Float16* H     = (_Float16*)(ws + 29425664);           // 16 MB  [8192 x 1024]
    float*    ladbf = (float*)   (ws + 46202880);           // 8 MB   [256 x 8192]

    float* out = (float*)d_out;
    float* lad = out + BATCH * FEAT;

    make_identity_f16<<<(BATCH * NTR + 255) / 256, 256, 0, stream>>>(inp, I16h, out);
    {   // W1t = W1^T (f16):  [256 x 1024] -> [1024 x 256]
        dim3 g(1024 / 32, 256 / 32);
        transpose_f32_to_f16<<<g, 256, 0, stream>>>(W1, W1t, 256, 1024, 1024);
    }
    {   // W2t = W2^T (f16):  [1024 x 12032] -> [12064 x 1024], pad zeroed
        dim3 g(NPAD / 32, 1024 / 32);
        transpose_f32_to_f16<<<g, 256, 0, stream>>>(W2, W2t, 1024, OUTDIM, NPAD);
    }
    dim3 g1(BATCH / 128, HIDDEN / 64);
    gemm1_relu<<<g1, 256, 0, stream>>>(I16h, W1t, b1, H);

    dim3 g2(BATCH / 128, NTR / 2);
    gemm2_spline<<<g2, 256, 0, stream>>>(H, W2t, b2, inp, out, ladbf);

    reduce_lad<<<BATCH / 256, 256, 0, stream>>>(ladbf, lad);
}
